// SparseAutoencoder_84963043050117
// MI455X (gfx1250) — matrix operand order repaired and checked by reading, not compiled, not run on hardware
//
#include <hip/hip_runtime.h>
#include <hip/hip_bf16.h>

// ---------------------------------------------------------------------------
// Problem constants (from reference)
// ---------------------------------------------------------------------------
#define BATCH   4096
#define D_IN    4096
#define H_DIM   32768
#define K_TOP   64
#define CAP     96        // compact top-k slots per row (>= K_TOP, tie slack)

// ---------------------------------------------------------------------------
// CDNA5 WMMA types
// ---------------------------------------------------------------------------
typedef __bf16 v16bf __attribute__((ext_vector_type(16)));
typedef float  v8f   __attribute__((ext_vector_type(8)));

struct Frag32 { uint4 a, b; };   // 32 bytes == one v16bf fragment

// round-to-nearest-even f32 -> bf16 pair packed into one dword
static __device__ __forceinline__ unsigned pack_bf16x2(float lo, float hi) {
    unsigned ul, uh;
    __builtin_memcpy(&ul, &lo, 4);
    __builtin_memcpy(&uh, &hi, 4);
    ul = (ul + 0x7FFFu + ((ul >> 16) & 1u)) >> 16;
    uh = (uh + 0x7FFFu + ((uh >> 16) & 1u)) >> 16;
    return (ul & 0xFFFFu) | (uh << 16);
}

static __device__ __forceinline__ float bf16lo_f32(unsigned u) {
    unsigned b = u << 16; float f; __builtin_memcpy(&f, &b, 4); return f;
}
static __device__ __forceinline__ float bf16hi_f32(unsigned u) {
    unsigned b = u & 0xFFFF0000u; float f; __builtin_memcpy(&f, &b, 4); return f;
}

// gfx1250 async DMA: global -> LDS, 16 bytes per lane (ASYNCcnt-tracked)
static __device__ __forceinline__ void async_cp16(unsigned lds_off,
                                                  unsigned long long gaddr) {
    asm volatile("global_load_async_to_lds_b128 %0, %1, off"
                 :: "v"(lds_off), "v"(gaddr) : "memory");
}

// ---------------------------------------------------------------------------
// flat f32 -> bf16 conversion (8 elements / thread)
// ---------------------------------------------------------------------------
__global__ __launch_bounds__(256)
void cvt_bf16(const float* __restrict__ in, unsigned* __restrict__ out, long n)
{
    long i = ((long)blockIdx.x * 256 + threadIdx.x) * 8;
    if (i >= n) return;
    float4 a = *(const float4*)(in + i);
    float4 b = *(const float4*)(in + i + 4);
    uint4 o;
    o.x = pack_bf16x2(a.x, a.y);
    o.y = pack_bf16x2(a.z, a.w);
    o.z = pack_bf16x2(b.x, b.y);
    o.w = pack_bf16x2(b.z, b.w);
    *(uint4*)(out + i / 2) = o;
}

// ---------------------------------------------------------------------------
// W_dec [D, H] f32  ->  W_decT [H, D] bf16   (64x64 LDS tiles)
// ---------------------------------------------------------------------------
__global__ __launch_bounds__(256)
void transpose_cvt(const float* __restrict__ W, unsigned* __restrict__ WT)
{
    __shared__ float tile[64][65];
    const int tid = threadIdx.x;
    const size_t h0 = (size_t)blockIdx.x * 64;
    const size_t d0 = (size_t)blockIdx.y * 64;

    #pragma unroll
    for (int p = 0; p < 4; ++p) {
        const int r = p * 16 + (tid >> 4);      // d offset
        const int c = (tid & 15) * 4;           // h offset
        float4 v = *(const float4*)&W[(d0 + r) * (size_t)H_DIM + h0 + c];
        tile[r][c] = v.x; tile[r][c+1] = v.y; tile[r][c+2] = v.z; tile[r][c+3] = v.w;
    }
    __syncthreads();

    #pragma unroll
    for (int p = 0; p < 2; ++p) {
        const int hh = p * 32 + (tid >> 3);     // h offset
        const int c8 = (tid & 7) * 8;           // d offset (8 bf16 = 16B)
        uint4 o;
        o.x = pack_bf16x2(tile[c8 + 0][hh], tile[c8 + 1][hh]);
        o.y = pack_bf16x2(tile[c8 + 2][hh], tile[c8 + 3][hh]);
        o.z = pack_bf16x2(tile[c8 + 4][hh], tile[c8 + 5][hh]);
        o.w = pack_bf16x2(tile[c8 + 6][hh], tile[c8 + 7][hh]);
        *(uint4*)&WT[((h0 + hh) * (size_t)D_IN + d0 + c8) / 2] = o;
    }
}

// ---------------------------------------------------------------------------
// bf16 NT GEMM, async double-buffered LDS staging, ONE barrier per K-step.
//   C[M,N] = act( A[M,K] * Bsrc[N,K]^T + bias[N] )
// Block tile 128x128, K-step 64 (two k=32 WMMA sub-steps -> 16 WMMA/wave
// between barriers).  blockIdx.x = M tile (fast-varying: Bsrc tile L2-hot,
// A L2-resident).
// ---------------------------------------------------------------------------
__global__ __launch_bounds__(256)
void gemm_bf16_async(const unsigned* __restrict__ A,   // bf16 pairs
                     const unsigned* __restrict__ Bsrc,
                     const float* __restrict__ bias,
                     float* __restrict__ C,
                     int M, int N, int K, int relu)
{
    // [buf][ Atile 128x128B | Btile 128x128B ] = 2 x 32 KB = 64 KB
    __shared__ __align__(16) unsigned SH[16384];

    const int tid  = threadIdx.x;
    const int lane = tid & 31;
    const int wave = tid >> 5;
    const int wm   = wave & 1;
    const int wn   = wave >> 1;

    const size_t tileM = (size_t)blockIdx.x * 128;
    const size_t tileN = (size_t)blockIdx.y * 128;

    v8f acc[4][2];
    const v8f vzero = {0.f,0.f,0.f,0.f,0.f,0.f,0.f,0.f};
    #pragma unroll
    for (int mi = 0; mi < 4; ++mi)
        #pragma unroll
        for (int ni = 0; ni < 2; ++ni) acc[mi][ni] = vzero;

    const unsigned ldsbase = (unsigned)(uintptr_t)(&SH[0]);

    // staging: per tile 128 rows x 128B -> 1024 chunks of 16B; thread owns 4.
    unsigned long long Ag[4], Bg[4];
    unsigned dA[4], dB[4];
    #pragma unroll
    for (int j = 0; j < 4; ++j) {
        const int c = tid + j * 256;
        const int r = c >> 3;              // row in tile
        const int o = (c & 7) * 16;        // byte offset in row
        Ag[j] = (unsigned long long)(uintptr_t)
                ((const char*)A + ((tileM + r) * (size_t)K) * 2 + o);
        Bg[j] = (unsigned long long)(uintptr_t)
                ((const char*)Bsrc + ((tileN + r) * (size_t)K) * 2 + o);
        dA[j] = ldsbase + r * 128 + o;
        dB[j] = dA[j] + 16384;             // B tile after A tile
    }

    const int klane = lane >> 4;
    const int nlane = lane & 15;

    // preload buffer 0 (k = 0)
    #pragma unroll
    for (int j = 0; j < 4; ++j) { async_cp16(dA[j], Ag[j]); async_cp16(dB[j], Bg[j]); }

    for (int kk = 0; kk < K; kk += 64) {
        const int buf = (kk >> 6) & 1;

        // current buffer's fill (issued one step ago) must have landed
        asm volatile("s_wait_asynccnt 0x0" ::: "memory");
        __syncthreads();   // everyone past previous step's reads of buf^1

        if (kk + 64 < K) { // prefetch next step into the other buffer
            const unsigned nb = (buf ^ 1) * 32768u;
            const unsigned long long koff = (unsigned long long)(kk + 64) * 2;
            #pragma unroll
            for (int j = 0; j < 4; ++j) {
                async_cp16(dA[j] + nb, Ag[j] + koff);
                async_cp16(dB[j] + nb, Bg[j] + koff);
            }
        }

        const unsigned* Ab = &SH[buf * 8192];
        const unsigned* Bb = &SH[buf * 8192 + 4096];

        #pragma unroll
        for (int s = 0; s < 2; ++s) {      // two k=32 sub-steps
            v16bf afrag[4];
            #pragma unroll
            for (int mi = 0; mi < 4; ++mi) {
                const int row = wm * 64 + mi * 16 + nlane;
                Frag32 f;
                f.a = *(const uint4*)&Ab[row * 32 + s * 16 + klane * 8];
                f.b = *(const uint4*)&Ab[row * 32 + s * 16 + klane * 8 + 4];
                afrag[mi] = __builtin_bit_cast(v16bf, f);
            }
            v16bf bfrag[2];
            #pragma unroll
            for (int ni = 0; ni < 2; ++ni) {
                const int col = wn * 32 + ni * 16 + nlane;
                Frag32 f = *(const Frag32*)&Bb[col * 32 + s * 16 + klane * 8];
                bfrag[ni] = __builtin_bit_cast(v16bf, f);
            }
            #pragma unroll
            for (int ni = 0; ni < 2; ++ni)
                #pragma unroll
                for (int mi = 0; mi < 4; ++mi)
                    acc[mi][ni] = __builtin_amdgcn_wmma_f32_16x16x32_bf16(
                        false, afrag[mi], false, bfrag[ni],
                        (short)0, acc[mi][ni], false, false);
        }
    }

    const int rhalf = lane >> 4;
    #pragma unroll
    for (int ni = 0; ni < 2; ++ni) {
        const size_t col = tileN + wn * 32 + ni * 16 + nlane;
        const float bv = bias[col];
        #pragma unroll
        for (int mi = 0; mi < 4; ++mi)
            #pragma unroll
            for (int r = 0; r < 8; ++r) {
                const size_t row = tileM + wm * 64 + mi * 16 + rhalf * 8 + r;
                float v = acc[mi][ni][r] + bv;
                if (relu) v = v > 0.0f ? v : 0.0f;
                C[row * (size_t)N + col] = v;
            }
    }
}

// ---------------------------------------------------------------------------
// FALLBACK: fused-conversion f32 GEMM (used when workspace is too small)
// ---------------------------------------------------------------------------
__global__ __launch_bounds__(256)
void gemm_nt_f32src(const float* __restrict__ A, const float* __restrict__ Bsrc,
                    const float* __restrict__ bias, float* __restrict__ C,
                    int M, int N, int K, int relu)
{
    __shared__ __align__(16) unsigned As[128][16];
    __shared__ __align__(16) unsigned Bs[128][16];

    const int tid = threadIdx.x, lane = tid & 31, wave = tid >> 5;
    const int wm = wave & 1, wn = wave >> 1;
    const size_t tileM = (size_t)blockIdx.x * 128;
    const size_t tileN = (size_t)blockIdx.y * 128;

    v8f acc[4][2];
    const v8f vzero = {0.f,0.f,0.f,0.f,0.f,0.f,0.f,0.f};
    #pragma unroll
    for (int mi = 0; mi < 4; ++mi)
        #pragma unroll
        for (int ni = 0; ni < 2; ++ni) acc[mi][ni] = vzero;

    const int ldrow = tid >> 1, ldhalf = tid & 1;
    const float* Ag = A    + (tileM + ldrow) * (size_t)K + ldhalf * 16;
    const float* Bg = Bsrc + (tileN + ldrow) * (size_t)K + ldhalf * 16;
    const int klane = lane >> 4, nlane = lane & 15;

    for (int kk = 0; kk < K; kk += 32) {
        __syncthreads();
        float4 a0 = *(const float4*)(Ag),     a1 = *(const float4*)(Ag + 4);
        float4 a2 = *(const float4*)(Ag + 8), a3 = *(const float4*)(Ag + 12);
        float4 b0 = *(const float4*)(Bg),     b1 = *(const float4*)(Bg + 4);
        float4 b2 = *(const float4*)(Bg + 8), b3 = *(const float4*)(Bg + 12);
        if (kk + 32 < K) { __builtin_prefetch(Ag + 32, 0, 0); __builtin_prefetch(Bg + 32, 0, 0); }
        Ag += 32; Bg += 32;
        unsigned* ap = &As[ldrow][ldhalf * 8];
        ap[0]=pack_bf16x2(a0.x,a0.y); ap[1]=pack_bf16x2(a0.z,a0.w);
        ap[2]=pack_bf16x2(a1.x,a1.y); ap[3]=pack_bf16x2(a1.z,a1.w);
        ap[4]=pack_bf16x2(a2.x,a2.y); ap[5]=pack_bf16x2(a2.z,a2.w);
        ap[6]=pack_bf16x2(a3.x,a3.y); ap[7]=pack_bf16x2(a3.z,a3.w);
        unsigned* bp = &Bs[ldrow][ldhalf * 8];
        bp[0]=pack_bf16x2(b0.x,b0.y); bp[1]=pack_bf16x2(b0.z,b0.w);
        bp[2]=pack_bf16x2(b1.x,b1.y); bp[3]=pack_bf16x2(b1.z,b1.w);
        bp[4]=pack_bf16x2(b2.x,b2.y); bp[5]=pack_bf16x2(b2.z,b2.w);
        bp[6]=pack_bf16x2(b3.x,b3.y); bp[7]=pack_bf16x2(b3.z,b3.w);
        __syncthreads();

        v16bf afrag[4]; v16bf bfrag[2];
        #pragma unroll
        for (int mi = 0; mi < 4; ++mi) {
            const int row = wm * 64 + mi * 16 + nlane;
            Frag32 f;
            f.a = *(const uint4*)&As[row][klane * 8];
            f.b = *(const uint4*)&As[row][klane * 8 + 4];
            afrag[mi] = __builtin_bit_cast(v16bf, f);
        }
        #pragma unroll
        for (int ni = 0; ni < 2; ++ni) {
            const int col = wn * 32 + ni * 16 + nlane;
            Frag32 f = *(const Frag32*)&Bs[col][klane * 8];
            bfrag[ni] = __builtin_bit_cast(v16bf, f);
        }
        #pragma unroll
        for (int ni = 0; ni < 2; ++ni)
            #pragma unroll
            for (int mi = 0; mi < 4; ++mi)
                acc[mi][ni] = __builtin_amdgcn_wmma_f32_16x16x32_bf16(
                    false, afrag[mi], false, bfrag[ni], (short)0, acc[mi][ni], false, false);
    }

    const int rhalf = lane >> 4;
    #pragma unroll
    for (int ni = 0; ni < 2; ++ni) {
        const size_t col = tileN + wn * 32 + ni * 16 + nlane;
        const float bv = bias[col];
        #pragma unroll
        for (int mi = 0; mi < 4; ++mi)
            #pragma unroll
            for (int r = 0; r < 8; ++r) {
                const size_t row = tileM + wm * 64 + mi * 16 + rhalf * 8 + r;
                float v = acc[mi][ni][r] + bv;
                if (relu) v = v > 0.0f ? v : 0.0f;
                C[row * (size_t)N + col] = v;
            }
    }
}

// ---------------------------------------------------------------------------
// Per-row top-K: 3-pass radix histogram threshold (bits monotone, post-ReLU),
// in-place masking of h, plus deterministic stable compaction of kept
// (val, idx) pairs into ws for the sparse decoder.
// ---------------------------------------------------------------------------
__global__ __launch_bounds__(1024)
void topk_row(float* __restrict__ h, int H, int Ktop,
              float* __restrict__ ovals, int* __restrict__ oidx,
              int* __restrict__ ocnt, int emit)
{
    __shared__ unsigned hist[8][256];
    __shared__ unsigned scan[1024];
    __shared__ unsigned s_prefix, s_remaining;

    const int tid = threadIdx.x;
    const int row = blockIdx.x;
    float* p = h + (size_t)row * (size_t)H;

    unsigned bits[32];
    #pragma unroll
    for (int i = 0; i < 32; ++i) {
        float v = p[i * 1024 + tid];
        __builtin_memcpy(&bits[i], &v, 4);
    }

    if (tid == 0) { s_prefix = 0u; s_remaining = (unsigned)Ktop; }
    const int hb = (tid >> 7) & 7;

    for (int pass = 0; pass < 3; ++pass) {
        const int shift = 24 - 8 * pass;
        for (int b = tid; b < 8 * 256; b += 1024) ((unsigned*)hist)[b] = 0u;
        __syncthreads();
        const unsigned prefix = s_prefix;
        #pragma unroll
        for (int i = 0; i < 32; ++i) {
            const unsigned u = bits[i];
            const bool in = (pass == 0) ? (u != 0u) : ((u >> (shift + 8)) == prefix);
            if (in) atomicAdd(&hist[hb][(u >> shift) & 0xFFu], 1u);
        }
        __syncthreads();
        if (tid < 256) {
            unsigned s = 0;
            #pragma unroll
            for (int c = 0; c < 8; ++c) s += hist[c][tid];
            hist[0][tid] = s;
        }
        __syncthreads();
        if (tid == 0) {
            const unsigned rem = s_remaining;
            unsigned cum = 0; int chosen = 0;
            for (int b = 255; b >= 0; --b) {
                const unsigned c = hist[0][b];
                if (cum + c >= rem) { chosen = b; break; }
                cum += c;
            }
            s_prefix = (prefix << 8) | (unsigned)chosen;
            s_remaining = rem - cum;
        }
        __syncthreads();
    }

    const unsigned T = s_prefix << 8;

    // in-place mask
    unsigned kept = 0;
    #pragma unroll
    for (int i = 0; i < 32; ++i) {
        unsigned u = bits[i];
        const bool keep = (u >= T) && (u != 0u);
        kept += keep ? 1u : 0u;
        unsigned m = keep ? u : 0u;
        float v; __builtin_memcpy(&v, &m, 4);
        p[i * 1024 + tid] = v;
    }

    if (!emit) return;

    // stable (deterministic) compaction: block-wide inclusive scan of counts
    scan[tid] = kept;
    __syncthreads();
    for (int off = 1; off < 1024; off <<= 1) {
        unsigned v = (tid >= off) ? scan[tid - off] : 0u;
        __syncthreads();
        scan[tid] += v;
        __syncthreads();
    }
    unsigned base = scan[tid] - kept;
    const unsigned total = scan[1023];

    #pragma unroll
    for (int i = 0; i < 32; ++i) {
        const unsigned u = bits[i];
        if ((u >= T) && (u != 0u)) {
            if (base < CAP) {
                float v; __builtin_memcpy(&v, &u, 4);
                ovals[(size_t)row * CAP + base] = v;
                oidx [(size_t)row * CAP + base] = i * 1024 + tid;
            }
            ++base;
        }
    }
    if (tid == 0) ocnt[row] = (int)(total < CAP ? total : CAP);
}

// ---------------------------------------------------------------------------
// Sparse decode: x_hat[b,:] = sum_k val_k * W_decT[idx_k, :] + b_dec
// W_decT bf16 [H, D] row-contiguous.  One block per batch row, 16 cols/thread.
// ---------------------------------------------------------------------------
__global__ __launch_bounds__(256)
void sparse_decode(const float* __restrict__ vals, const int* __restrict__ idx,
                   const int* __restrict__ cnt, const unsigned* __restrict__ WT,
                   const float* __restrict__ b_dec, float* __restrict__ x_hat)
{
    const int tid = threadIdx.x;
    const int row = blockIdx.x;
    const int col0 = tid * 16;

    float acc[16];
    #pragma unroll
    for (int j = 0; j < 16; j += 4) {
        float4 b = *(const float4*)&b_dec[col0 + j];
        acc[j] = b.x; acc[j+1] = b.y; acc[j+2] = b.z; acc[j+3] = b.w;
    }

    const int n = cnt[row];
    for (int k = 0; k < n; ++k) {
        const float v = vals[(size_t)row * CAP + k];
        const int   id = idx[(size_t)row * CAP + k];
        const unsigned* wr = WT + (size_t)id * (D_IN / 2) + col0 / 2;
        uint4 w0 = *(const uint4*)(wr);
        uint4 w1 = *(const uint4*)(wr + 4);
        const unsigned wu[8] = {w0.x, w0.y, w0.z, w0.w, w1.x, w1.y, w1.z, w1.w};
        #pragma unroll
        for (int j = 0; j < 8; ++j) {
            acc[2*j]   += v * bf16lo_f32(wu[j]);
            acc[2*j+1] += v * bf16hi_f32(wu[j]);
        }
    }

    float* out = x_hat + (size_t)row * D_IN + col0;
    #pragma unroll
    for (int j = 0; j < 16; j += 4) {
        float4 o = {acc[j], acc[j+1], acc[j+2], acc[j+3]};
        *(float4*)(out + j) = o;
    }
}

// ---------------------------------------------------------------------------
// Launch
// d_in: x, W_enc, b_enc, W_dec, b_dec    d_out: [h | x_hat]
// ws layout: x_bf | W_enc_bf | W_decT_bf | vals | idx | cnt
// ---------------------------------------------------------------------------
extern "C" void kernel_launch(void* const* d_in, const int* in_sizes, int n_in,
                              void* d_out, int out_size, void* d_ws, size_t ws_size,
                              hipStream_t stream) {
    const float* x     = (const float*)d_in[0];
    const float* W_enc = (const float*)d_in[1];
    const float* b_enc = (const float*)d_in[2];
    const float* W_dec = (const float*)d_in[3];
    const float* b_dec = (const float*)d_in[4];

    float* h     = (float*)d_out;
    float* x_hat = h + (size_t)BATCH * (size_t)H_DIM;

    const size_t sz_xb  = (size_t)BATCH * D_IN * 2;
    const size_t sz_web = (size_t)H_DIM * D_IN * 2;
    const size_t sz_wdt = (size_t)H_DIM * D_IN * 2;
    const size_t sz_val = (size_t)BATCH * CAP * 4;
    const size_t sz_idx = (size_t)BATCH * CAP * 4;
    const size_t sz_cnt = (size_t)BATCH * 4;
    const size_t need = sz_xb + sz_web + sz_wdt + sz_val + sz_idx + sz_cnt;

    dim3 blk256(256);

    if (ws_size >= need) {
        char* w = (char*)d_ws;
        unsigned* xb  = (unsigned*)(w);
        unsigned* web = (unsigned*)(w + sz_xb);
        unsigned* wdt = (unsigned*)(w + sz_xb + sz_web);
        float*    pv  = (float*)   (w + sz_xb + sz_web + sz_wdt);
        int*      pi  = (int*)     (w + sz_xb + sz_web + sz_wdt + sz_val);
        int*      pc  = (int*)     (w + sz_xb + sz_web + sz_wdt + sz_val + sz_idx);

        // pre-convert operands to bf16 (one-time, memory-bound)
        cvt_bf16<<<(unsigned)(((size_t)BATCH * D_IN / 8 + 255) / 256), blk256, 0, stream>>>(
            x, xb, (long)BATCH * D_IN);
        cvt_bf16<<<(unsigned)(((size_t)H_DIM * D_IN / 8 + 255) / 256), blk256, 0, stream>>>(
            W_enc, web, (long)H_DIM * D_IN);
        transpose_cvt<<<dim3(H_DIM / 64, D_IN / 64), blk256, 0, stream>>>(W_dec, wdt);

        // encoder (M-tile fast-varying: W_enc streamed once, x L2-resident)
        gemm_bf16_async<<<dim3(BATCH / 128, H_DIM / 128), blk256, 0, stream>>>(
            xb, web, b_enc, h, BATCH, H_DIM, D_IN, 1);

        // top-k mask + deterministic compaction
        topk_row<<<BATCH, 1024, 0, stream>>>(h, H_DIM, K_TOP, pv, pi, pc, 1);

        // sparse decode
        sparse_decode<<<BATCH, blk256, 0, stream>>>(pv, pi, pc, wdt, b_dec, x_hat);
    } else {
        // fallback: fused-conversion GEMMs, dense decode
        gemm_nt_f32src<<<dim3(BATCH / 128, H_DIM / 128), blk256, 0, stream>>>(
            x, W_enc, b_enc, h, BATCH, H_DIM, D_IN, 1);
        topk_row<<<BATCH, 1024, 0, stream>>>(h, H_DIM, K_TOP,
                                             nullptr, nullptr, nullptr, 0);
        gemm_nt_f32src<<<dim3(BATCH / 128, D_IN / 128), blk256, 0, stream>>>(
            h, W_dec, b_dec, x_hat, BATCH, D_IN, H_DIM, 0);
    }
}
